// MyGCN_27917287424649
// MI455X (gfx1250) — matrix operand order repaired and checked by reading, not compiled, not run on hardware
//
#include <hip/hip_runtime.h>
#include <hip/hip_bf16.h>

typedef __attribute__((ext_vector_type(16))) _Float16 v16h;
typedef __attribute__((ext_vector_type(8)))  _Float16 v8h;
typedef __attribute__((ext_vector_type(8)))  float    v8f;

union F16x16 { v16h v; v8h h[2]; };

#define GN      4096
#define GNFEAT  1280
#define GNHID   512
#define GH      8
#define GHD     64
#define GDFF    2048
#define GE      65536
#define GB      16

// ---- epilogue flags ----
#define EPI_F32  1
#define EPI_F16  2
#define EPI_RELU 4
#define EPI_BIAS 8
#define EPI_QKV  16

// =====================================================================
// f32 -> f16 convert
__global__ void k_cvt(const float* __restrict__ x, _Float16* __restrict__ y, long n) {
  long i = (long)blockIdx.x * blockDim.x + threadIdx.x;
  if (i < n) y[i] = (_Float16)x[i];
}

// degree accumulation
__global__ void k_deg(const int* __restrict__ src, const int* __restrict__ dst,
                      float* __restrict__ dout, float* __restrict__ din, int E) {
  int e = blockIdx.x * blockDim.x + threadIdx.x;
  if (e < E) {
    atomicAdd(&dout[src[e]], 1.0f);
    atomicAdd(&din[dst[e]], 1.0f);
  }
}

// y[row,c] = x[row,c] * max(deg[row],1)^-1/2   (f32 -> f16)
__global__ void k_scale(const float* __restrict__ x, const float* __restrict__ deg,
                        _Float16* __restrict__ y, int cols, long total) {
  long i = (long)blockIdx.x * blockDim.x + threadIdx.x;
  if (i < total) {
    int row = (int)(i / cols);
    float d = deg[row]; d = d < 1.0f ? 1.0f : d;
    y[i] = (_Float16)(x[i] * rsqrtf(d));
  }
}

// scatter-add over edges: agg[dst] += h[src]   (512 channels)
__global__ __launch_bounds__(128)
void k_agg(const int* __restrict__ src, const int* __restrict__ dst,
           const float* __restrict__ h, float* __restrict__ agg) {
  int e = blockIdx.x;
  long s = (long)src[e] * GNHID, d = (long)dst[e] * GNHID;
  for (int c = threadIdx.x; c < GNHID; c += 128)
    atomicAdd(&agg[d + c], h[s + c]);
}

// out = relu(agg * max(deg_in,1)^-1/2 + b); optional f16 copy
__global__ void k_post(const float* __restrict__ agg, const float* __restrict__ din,
                       const float* __restrict__ b, float* __restrict__ of,
                       _Float16* __restrict__ oh, long total) {
  long i = (long)blockIdx.x * blockDim.x + threadIdx.x;
  if (i < total) {
    int row = (int)(i >> 9), c = (int)(i & 511);
    float d = din[row]; d = d < 1.0f ? 1.0f : d;
    float v = agg[i] * rsqrtf(d) + b[c];
    v = fmaxf(v, 0.0f);
    of[i] = v;
    if (oh) oh[i] = (_Float16)v;
  }
}

// y = LayerNorm(x + dlt) * g + b ; one wave per 512-wide row
__global__ __launch_bounds__(256)
void k_ln(const float* __restrict__ x, const float* __restrict__ dlt,
          const float* __restrict__ g, const float* __restrict__ b,
          float* __restrict__ yf, _Float16* __restrict__ yh) {
  int lane = threadIdx.x & 31, w = threadIdx.x >> 5;
  long row = (long)blockIdx.x * 8 + w;
  const float* px = x + row * GNHID;
  const float* pd = dlt + row * GNHID;
  float v[16]; float s = 0.f;
#pragma unroll
  for (int t = 0; t < 16; ++t) { v[t] = px[t * 32 + lane] + pd[t * 32 + lane]; s += v[t]; }
#pragma unroll
  for (int o = 16; o >= 1; o >>= 1) s += __shfl_xor(s, o);
  float mu = s * (1.0f / GNHID);
  float vs = 0.f;
#pragma unroll
  for (int t = 0; t < 16; ++t) { float d = v[t] - mu; vs += d * d; }
#pragma unroll
  for (int o = 16; o >= 1; o >>= 1) vs += __shfl_xor(vs, o);
  float rs = rsqrtf(vs * (1.0f / GNHID) + 1e-5f);
#pragma unroll
  for (int t = 0; t < 16; ++t) {
    int c = t * 32 + lane;
    float y = (v[t] - mu) * rs * g[c] + b[c];
    yf[row * GNHID + c] = y;
    if (yh) yh[row * GNHID + c] = (_Float16)y;
  }
}

// =====================================================================
// Tiled WMMA GEMM: C[M,N] = A[M,K] @ B[N,K]^T (+bias, relu, f16/f32/qkv epilogue)
// 256 threads = 8 waves arranged 2(M) x 4(N); wave owns 64x64 (4x4 WMMA tiles)
// -> 16 WMMAs per 8 fragment loads per K-step. Grid: (N/256, M/128)
__global__ __launch_bounds__(256)
void k_gemm(const _Float16* __restrict__ A, const _Float16* __restrict__ B,
            const float* __restrict__ bias, int M, int N, int K, int flags,
            float* __restrict__ Cf, _Float16* __restrict__ Ch,
            _Float16* __restrict__ Qh, _Float16* __restrict__ Kh, _Float16* __restrict__ Vt) {
  const int lane = threadIdx.x & 31;
  const int w = threadIdx.x >> 5;
  const int wm = w >> 2, wn = w & 3;
  const int mbase = blockIdx.y * 128 + wm * 64;
  const int nbase = blockIdx.x * 256 + wn * 64;
  const int l16 = lane & 15, hl = lane >> 4;

  v8f acc[4][4] = {};
  for (int kt = 0; kt < K; kt += 32) {
    F16x16 a[4], bfr[4];
#pragma unroll
    for (int i = 0; i < 4; ++i) {
      const _Float16* pa = A + (size_t)(mbase + i * 16 + l16) * K + kt;
      a[i].h[0] = *(const v8h*)(pa + hl * 16);
      a[i].h[1] = *(const v8h*)(pa + hl * 16 + 8);
    }
#pragma unroll
    for (int j = 0; j < 4; ++j) {
      const _Float16* pb = B + (size_t)(nbase + j * 16 + l16) * K + kt + hl * 16;
      bfr[j].h[0] = *(const v8h*)(pb);
      bfr[j].h[1] = *(const v8h*)(pb + 8);
    }
#pragma unroll
    for (int i = 0; i < 4; ++i)
#pragma unroll
      for (int j = 0; j < 4; ++j)
        acc[i][j] = __builtin_amdgcn_wmma_f32_16x16x32_f16(
            false, a[i].v, false, bfr[j].v, (short)0, acc[i][j], false, false);
  }

#pragma unroll
  for (int i = 0; i < 4; ++i)
#pragma unroll
    for (int j = 0; j < 4; ++j)
#pragma unroll
      for (int r = 0; r < 8; ++r) {
        int m = mbase + i * 16 + r + 8 * hl;
        int n = nbase + j * 16 + l16;
        float v = acc[i][j][r];
        if (flags & EPI_BIAS) v += bias[n];
        if (flags & EPI_RELU) v = fmaxf(v, 0.0f);
        if (flags & EPI_QKV) {
          if (n < GNHID) {               // Q, pre-scaled by 1/sqrt(hd)
            int hh = n >> 6, d = n & 63;
            Qh[((size_t)hh * GN + m) * GHD + d] = (_Float16)(v * 0.125f);
          } else if (n < 2 * GNHID) {    // K row-major [H][S][hd]
            int c = n - GNHID, hh = c >> 6, d = c & 63;
            Kh[((size_t)hh * GN + m) * GHD + d] = (_Float16)v;
          } else {                       // V transposed [H][hd][S]
            int c = n - 2 * GNHID, hh = c >> 6, d = c & 63;
            Vt[((size_t)hh * GHD + d) * GN + m] = (_Float16)v;
          }
        } else {
          if (flags & EPI_F32) Cf[(size_t)m * N + n] = v;
          if (flags & EPI_F16) Ch[(size_t)m * N + n] = (_Float16)v;
        }
      }
}

// =====================================================================
// Flash attention: one wave per (head, 32-row query block). S=4096, hd=64.
// K/V fragments are reused across the two 16-row query subtiles:
// 16 WMMAs per 8 fragment loads per 32-key step.
// Grid: 256 blocks x 128 threads (4 waves).
__global__ __launch_bounds__(128)
void k_attn(const _Float16* __restrict__ Q, const _Float16* __restrict__ Kt,
            const _Float16* __restrict__ Vt, _Float16* __restrict__ ctx) {
  __shared__ __align__(16) _Float16 plds[4][2][16][32];
  const int lane = threadIdx.x & 31, w = threadIdx.x >> 5;
  const int gid = blockIdx.x * 4 + w;     // 0..1023
  const int head = gid >> 7;              // 128 query-pairs per head
  const int qp = gid & 127;               // 32-row query block index
  const int l16 = lane & 15, hl = lane >> 4;
  const _Float16* Qh = Q + (size_t)head * GN * GHD;
  const _Float16* Kh = Kt + (size_t)head * GN * GHD;
  const _Float16* Vh = Vt + (size_t)head * GHD * GN;

  F16x16 qf[2][2];
#pragma unroll
  for (int qs = 0; qs < 2; ++qs) {
    const _Float16* pq = Qh + (size_t)(qp * 32 + qs * 16 + l16) * GHD;
#pragma unroll
    for (int f = 0; f < 2; ++f) {
      qf[qs][f].h[0] = *(const v8h*)(pq + f * 32 + hl * 16);
      qf[qs][f].h[1] = *(const v8h*)(pq + f * 32 + hl * 16 + 8);
    }
  }
  v8f O[2][4] = {};
  float mr[2][8], lr[2][8];
#pragma unroll
  for (int qs = 0; qs < 2; ++qs)
#pragma unroll
    for (int r = 0; r < 8; ++r) { mr[qs][r] = -1e30f; lr[qs][r] = 0.0f; }

  for (int kk = 0; kk < GN; kk += 32) {
    v8f sc[2][2] = {};
#pragma unroll
    for (int half = 0; half < 2; ++half) {
      F16x16 kb0, kb1;
      const _Float16* pk0 = Kh + (size_t)(kk + l16) * GHD + half * 32 + hl * 16;
      const _Float16* pk1 = Kh + (size_t)(kk + 16 + l16) * GHD + half * 32 + hl * 16;
      kb0.h[0] = *(const v8h*)pk0; kb0.h[1] = *(const v8h*)(pk0 + 8);
      kb1.h[0] = *(const v8h*)pk1; kb1.h[1] = *(const v8h*)(pk1 + 8);
#pragma unroll
      for (int qs = 0; qs < 2; ++qs) {
        sc[qs][0] = __builtin_amdgcn_wmma_f32_16x16x32_f16(
            false, qf[qs][half].v, false, kb0.v, (short)0, sc[qs][0], false, false);
        sc[qs][1] = __builtin_amdgcn_wmma_f32_16x16x32_f16(
            false, qf[qs][half].v, false, kb1.v, (short)0, sc[qs][1], false, false);
      }
    }
#pragma unroll
    for (int qs = 0; qs < 2; ++qs) {
      float alpha[8];
#pragma unroll
      for (int r = 0; r < 8; ++r) {
        float t = fmaxf(sc[qs][0][r], sc[qs][1][r]);
        t = fmaxf(t, __shfl_xor(t, 1)); t = fmaxf(t, __shfl_xor(t, 2));
        t = fmaxf(t, __shfl_xor(t, 4)); t = fmaxf(t, __shfl_xor(t, 8));
        float mn = fmaxf(mr[qs][r], t);
        alpha[r] = __expf(mr[qs][r] - mn);
        float p0 = __expf(sc[qs][0][r] - mn), p1 = __expf(sc[qs][1][r] - mn);
        float rs = p0 + p1;
        rs += __shfl_xor(rs, 1); rs += __shfl_xor(rs, 2);
        rs += __shfl_xor(rs, 4); rs += __shfl_xor(rs, 8);
        lr[qs][r] = lr[qs][r] * alpha[r] + rs;
        mr[qs][r] = mn;
        int m = r + 8 * hl;
        plds[w][qs][m][l16] = (_Float16)p0;        // P transpose staging
        plds[w][qs][m][16 + l16] = (_Float16)p1;
      }
#pragma unroll
      for (int dt = 0; dt < 4; ++dt)
#pragma unroll
        for (int r = 0; r < 8; ++r) O[qs][dt][r] *= alpha[r];
    }
    F16x16 pf[2];   // A-fragments of P (same-wave DS ops are in-order)
#pragma unroll
    for (int qs = 0; qs < 2; ++qs) {
      pf[qs].h[0] = *(const v8h*)&plds[w][qs][l16][hl * 16];
      pf[qs].h[1] = *(const v8h*)&plds[w][qs][l16][hl * 16 + 8];
    }
#pragma unroll
    for (int dt = 0; dt < 4; ++dt) {
      const _Float16* pv = Vh + (size_t)(dt * 16 + l16) * GN + kk + hl * 16;
      F16x16 vb; vb.h[0] = *(const v8h*)pv; vb.h[1] = *(const v8h*)(pv + 8);
#pragma unroll
      for (int qs = 0; qs < 2; ++qs)
        O[qs][dt] = __builtin_amdgcn_wmma_f32_16x16x32_f16(
            false, pf[qs].v, false, vb.v, (short)0, O[qs][dt], false, false);
    }
  }
#pragma unroll
  for (int qs = 0; qs < 2; ++qs)
#pragma unroll
    for (int dt = 0; dt < 4; ++dt)
#pragma unroll
      for (int r = 0; r < 8; ++r) {
        int m = qp * 32 + qs * 16 + r + 8 * hl;
        int n = head * GHD + dt * 16 + l16;
        ctx[(size_t)m * GNHID + n] = (_Float16)(O[qs][dt][r] / lr[qs][r]);
      }
}

// =====================================================================
// segment means (16 contiguous graphs of 256 nodes) -> hg = mean1 * mean2
__global__ __launch_bounds__(512)
void k_mean(const float* __restrict__ h1, const float* __restrict__ h2, float* __restrict__ hg) {
  int g = blockIdx.x, c = threadIdx.x;
  float s1 = 0.f, s2 = 0.f;
  for (int r = 0; r < 256; ++r) {
    s1 += h1[(size_t)(g * 256 + r) * GNHID + c];
    s2 += h2[(size_t)(g * 256 + r) * GNHID + c];
  }
  hg[g * GNHID + c] = (s1 * (1.0f / 256.0f)) * (s2 * (1.0f / 256.0f));
}

// tiny FC (no activation, matches reference)
__global__ void k_fc(const float* __restrict__ x, const float* __restrict__ W,
                     const float* __restrict__ b, float* __restrict__ y,
                     int M, int N, int K) {
  int t = blockIdx.x * blockDim.x + threadIdx.x;
  if (t < M * N) {
    int m = t / N, n = t % N;
    float s = b[n];
    for (int k = 0; k < K; ++k) s += x[(size_t)m * K + k] * W[(size_t)n * K + k];
    y[t] = s;
  }
}

// final [16,128] @ [2,128]^T + softmax
__global__ void k_head3(const float* __restrict__ x, const float* __restrict__ W,
                        const float* __restrict__ b, float* __restrict__ out) {
  int m = threadIdx.x;
  if (m < GB) {
    float z0 = b[0], z1 = b[1];
    for (int k = 0; k < 128; ++k) {
      float xv = x[m * 128 + k];
      z0 += xv * W[k];
      z1 += xv * W[128 + k];
    }
    float mx = fmaxf(z0, z1);
    float e0 = __expf(z0 - mx), e1 = __expf(z1 - mx);
    float s = e0 + e1;
    out[m * 2 + 0] = e0 / s;
    out[m * 2 + 1] = e1 / s;
  }
}

// =====================================================================
extern "C" void kernel_launch(void* const* d_in, const int* in_sizes, int n_in,
                              void* d_out, int out_size, void* d_ws, size_t ws_size,
                              hipStream_t stream) {
  const float* fea[2] = {(const float*)d_in[0], (const float*)d_in[1]};
  const int* src[2]   = {(const int*)d_in[2], (const int*)d_in[4]};
  const int* dst[2]   = {(const int*)d_in[3], (const int*)d_in[5]};
  const float* W1  = (const float*)d_in[9];
  const float* b1  = (const float*)d_in[10];
  const float* W2  = (const float*)d_in[11];
  const float* b2  = (const float*)d_in[12];
  const float* ipw = (const float*)d_in[13];
  const float* ipb = (const float*)d_in[14];
  const float* opw = (const float*)d_in[15];
  const float* opb = (const float*)d_in[16];
  const float* ln1g = (const float*)d_in[17];
  const float* ln1b = (const float*)d_in[18];
  const float* ln2g = (const float*)d_in[19];
  const float* ln2b = (const float*)d_in[20];
  const float* ff1w = (const float*)d_in[21];
  const float* ff1b = (const float*)d_in[22];
  const float* ff2w = (const float*)d_in[23];
  const float* ff2b = (const float*)d_in[24];
  const float* l1w = (const float*)d_in[25];
  const float* l1b = (const float*)d_in[26];
  const float* l2w = (const float*)d_in[27];
  const float* l2b = (const float*)d_in[28];
  const float* l3w = (const float*)d_in[29];
  const float* l3b = (const float*)d_in[30];

  size_t off = 0;
  auto alloc = [&](size_t bytes) -> void* {
    void* p = (char*)d_ws + off;
    off += (bytes + 255) & ~(size_t)255;
    return p;
  };

  // weight f16 copies
  _Float16* W1h  = (_Float16*)alloc((size_t)GNHID * GNFEAT * 2);
  _Float16* W2h  = (_Float16*)alloc((size_t)GNHID * GNHID * 2);
  _Float16* ipwh = (_Float16*)alloc((size_t)3 * GNHID * GNHID * 2);
  _Float16* opwh = (_Float16*)alloc((size_t)GNHID * GNHID * 2);
  _Float16* ff1h = (_Float16*)alloc((size_t)GDFF * GNHID * 2);
  _Float16* ff2h = (_Float16*)alloc((size_t)GNHID * GDFF * 2);
  auto cvt = [&](const float* s, _Float16* d, long n) {
    k_cvt<<<(int)((n + 255) / 256), 256, 0, stream>>>(s, d, n);
  };
  cvt(W1, W1h, (long)GNHID * GNFEAT);
  cvt(W2, W2h, (long)GNHID * GNHID);
  cvt(ipw, ipwh, (long)3 * GNHID * GNHID);
  cvt(opw, opwh, (long)GNHID * GNHID);
  cvt(ff1w, ff1h, (long)GDFF * GNHID);
  cvt(ff2w, ff2h, (long)GNHID * GDFF);

  const long NH = (long)GN * GNHID;       // 2M elements
  float* hout[2];

  for (int s = 0; s < 2; ++s) {
    float*     dout = (float*)alloc(GN * 4);
    float*     din  = (float*)alloc(GN * 4);
    _Float16*  feah = (_Float16*)alloc((size_t)GN * GNFEAT * 2);
    float*     gc1  = (float*)alloc(NH * 4);
    float*     agg1 = (float*)alloc(NH * 4);
    float*     h1f  = (float*)alloc(NH * 4);
    _Float16*  h1h  = (_Float16*)alloc(NH * 2);
    _Float16*  Qb   = (_Float16*)alloc(NH * 2);
    _Float16*  Kb   = (_Float16*)alloc(NH * 2);
    _Float16*  Vtb  = (_Float16*)alloc(NH * 2);
    _Float16*  ctxb = (_Float16*)alloc(NH * 2);
    float*     ao   = (float*)alloc(NH * 4);
    float*     x1   = (float*)alloc(NH * 4);
    _Float16*  x1h  = (_Float16*)alloc(NH * 2);
    _Float16*  ffh  = (_Float16*)alloc((size_t)GN * GDFF * 2);
    float*     ffo  = (float*)alloc(NH * 4);
    float*     x2   = (float*)alloc(NH * 4);
    _Float16*  x2sh = (_Float16*)alloc(NH * 2);
    float*     gc2  = (float*)alloc(NH * 4);
    float*     agg2 = (float*)alloc(NH * 4);
    float*     ho   = (float*)alloc(NH * 4);
    hout[s] = ho;

    // degrees
    hipMemsetAsync(dout, 0, GN * 4, stream);
    hipMemsetAsync(din, 0, GN * 4, stream);
    k_deg<<<GE / 256, 256, 0, stream>>>(src[s], dst[s], dout, din, GE);

    // GraphConv1: scale -> GEMM -> scatter -> scale+bias+relu
    k_scale<<<(int)(((long)GN * GNFEAT + 255) / 256), 256, 0, stream>>>(
        fea[s], dout, feah, GNFEAT, (long)GN * GNFEAT);
    k_gemm<<<dim3(GNHID / 256, GN / 128), 256, 0, stream>>>(
        feah, W1h, nullptr, GN, GNHID, GNFEAT, EPI_F32, gc1, nullptr, nullptr, nullptr, nullptr);
    hipMemsetAsync(agg1, 0, NH * 4, stream);
    k_agg<<<GE, 128, 0, stream>>>(src[s], dst[s], gc1, agg1);
    k_post<<<(int)((NH + 255) / 256), 256, 0, stream>>>(agg1, din, b1, h1f, h1h, NH);

    // Encoder: QKV projection (writes Q scaled, K, V^T in f16)
    k_gemm<<<dim3(3 * GNHID / 256, GN / 128), 256, 0, stream>>>(
        h1h, ipwh, ipb, GN, 3 * GNHID, GNHID, EPI_BIAS | EPI_QKV,
        nullptr, nullptr, Qb, Kb, Vtb);
    // flash attention
    k_attn<<<256, 128, 0, stream>>>(Qb, Kb, Vtb, ctxb);
    // out projection
    k_gemm<<<dim3(GNHID / 256, GN / 128), 256, 0, stream>>>(
        ctxb, opwh, opb, GN, GNHID, GNHID, EPI_BIAS | EPI_F32, ao, nullptr, nullptr, nullptr, nullptr);
    // residual + LN1
    k_ln<<<GN / 8, 256, 0, stream>>>(h1f, ao, ln1g, ln1b, x1, x1h);
    // FF
    k_gemm<<<dim3(GDFF / 256, GN / 128), 256, 0, stream>>>(
        x1h, ff1h, ff1b, GN, GDFF, GNHID, EPI_BIAS | EPI_RELU | EPI_F16,
        nullptr, ffh, nullptr, nullptr, nullptr);
    k_gemm<<<dim3(GNHID / 256, GN / 128), 256, 0, stream>>>(
        ffh, ff2h, ff2b, GN, GNHID, GDFF, EPI_BIAS | EPI_F32, ffo, nullptr, nullptr, nullptr, nullptr);
    // residual + LN2
    k_ln<<<GN / 8, 256, 0, stream>>>(x1, ffo, ln2g, ln2b, x2, nullptr);

    // GraphConv2
    k_scale<<<(int)((NH + 255) / 256), 256, 0, stream>>>(x2, dout, x2sh, GNHID, NH);
    k_gemm<<<dim3(GNHID / 256, GN / 128), 256, 0, stream>>>(
        x2sh, W2h, nullptr, GN, GNHID, GNHID, EPI_F32, gc2, nullptr, nullptr, nullptr, nullptr);
    hipMemsetAsync(agg2, 0, NH * 4, stream);
    k_agg<<<GE, 128, 0, stream>>>(src[s], dst[s], gc2, agg2);
    k_post<<<(int)((NH + 255) / 256), 256, 0, stream>>>(agg2, din, b2, ho, nullptr, NH);
  }

  // readout + head
  float* hg = (float*)alloc((size_t)GB * GNHID * 4);
  float* o1 = (float*)alloc((size_t)GB * 512 * 4);
  float* o2 = (float*)alloc((size_t)GB * 128 * 4);
  k_mean<<<GB, GNHID, 0, stream>>>(hout[0], hout[1], hg);
  k_fc<<<(GB * 512 + 255) / 256, 256, 0, stream>>>(hg, l1w, l1b, o1, GB, 512, GNHID);
  k_fc<<<(GB * 128 + 255) / 256, 256, 0, stream>>>(o1, l2w, l2b, o2, GB, 128, 512);
  k_head3<<<1, 32, 0, stream>>>(o2, l3w, l3b, (float*)d_out);
}
